// Decoder_24867860644480
// MI455X (gfx1250) — matrix operand order repaired and checked by reading, not compiled, not run on hardware
//
#include <hip/hip_runtime.h>
#include <stdint.h>

// Problem dims (fixed by the reference)
#define VSZ    30000
#define EDIM   1024
#define SHDIM  2048
#define HDIM   1024
#define BDIM   64
#define TDIM   50
#define PADTOK 10003
#define H2     2048
#define G3     6144   // 3*H2
#define ROWS   (TDIM * BDIM)  // 3200 (t-major, row = t*64 + b)

typedef __attribute__((ext_vector_type(16))) __bf16 v16bf;
typedef __attribute__((ext_vector_type(8)))  float  v8f;

union FragU { uint4 u[2]; v16bf v; };

__device__ __forceinline__ uint16_t f2bf(float f) {
  union { float f; uint32_t u; } a; a.f = f;
  uint32_t r = a.u + 0x7FFFu + ((a.u >> 16) & 1u);  // RNE
  return (uint16_t)(r >> 16);
}
__device__ __forceinline__ float bf2f(uint16_t h) {
  union { uint32_t u; float f; } a; a.u = ((uint32_t)h) << 16;
  return a.f;
}
__device__ __forceinline__ v8f vzero8() {
  v8f z;
#pragma unroll
  for (int i = 0; i < 8; ++i) z[i] = 0.0f;
  return z;
}

// ISA 16-bit A layout (16x32): lane<16 -> M=lane, VGPR0-3: K=(8h)+0..7, VGPR4-7: K=16+(8h)+0..7
__device__ __forceinline__ v16bf frag_a(const uint16_t* A, int lda, int m0, int k0) {
  int lane = threadIdx.x & 31;
  int r = lane & 15, h = lane >> 4;
  const uint16_t* p = A + (size_t)(m0 + r) * lda + k0 + h * 8;
  FragU f;
  f.u[0] = *(const uint4*)(p);        // K: k0+8h .. +7
  f.u[1] = *(const uint4*)(p + 16);   // K: k0+16+8h .. +7
  return f.v;
}
// ISA 16-bit B layout (32x16) from row-major W (we compute A·Wᵀ): lane half h covers K=k0+8h..+7 and K=k0+16+8h..+7
__device__ __forceinline__ v16bf frag_b(const uint16_t* Bw, int ldb, int n0, int k0) {
  int lane = threadIdx.x & 31;
  int r = lane & 15, h = lane >> 4;
  const uint16_t* p = Bw + (size_t)(n0 + r) * ldb + k0 + h * 8;
  FragU f;
  f.u[0] = *(const uint4*)(p);
  f.u[1] = *(const uint4*)(p + 16);
  return f.v;
}
__device__ __forceinline__ v8f wmma_bf16(v16bf a, v16bf b, v8f c) {
  return __builtin_amdgcn_wmma_f32_16x16x32_bf16(false, a, false, b, (short)0, c,
                                                 false, false);
}

// Wave tile: 32M x (NT*16)N. Block: 8 waves -> 32M x (NT*128)N.
// Branch-free inner loop: ragged N handled by clamping the tile start to the
// last valid tile (Ncols is always a multiple of 16). WMMA executes with full
// EXEC (ISA requirement); the epilogue's store guard drops the clamped tiles.
template<int NT>
__device__ __forceinline__ void gemm_core(const uint16_t* __restrict__ A,
                                          const uint16_t* __restrict__ Bw,
                                          int K, int Ncols, int m0, int n0base,
                                          v8f acc[NT][2]) {
  int n0c[NT];
#pragma unroll
  for (int j = 0; j < NT; ++j) {
    int n0 = n0base + j * 16;
    n0c[j] = (n0 < Ncols) ? n0 : (Ncols - 16);
    acc[j][0] = vzero8();
    acc[j][1] = vzero8();
  }
#pragma unroll 2
  for (int k0 = 0; k0 < K; k0 += 32) {
    v16bf a0 = frag_a(A, K, m0, k0);
    v16bf a1 = frag_a(A, K, m0 + 16, k0);
#pragma unroll
    for (int j = 0; j < NT; ++j) {
      v16bf b = frag_b(Bw, K, n0c[j], k0);
      acc[j][0] = wmma_bf16(a0, b, acc[j][0]);
      acc[j][1] = wmma_bf16(a1, b, acc[j][1]);
    }
  }
}

// ---------------- elementwise / gather kernels ----------------
__global__ void k_convert(const float* __restrict__ src, uint16_t* __restrict__ dst, size_t n) {
  size_t i = (size_t)blockIdx.x * blockDim.x + threadIdx.x;
  size_t stride = (size_t)gridDim.x * blockDim.x;
  for (; i < n; i += stride) dst[i] = f2bf(src[i]);
}

// emb_bf[row=t*64+b][e] = bf16(emb_W[x[b,t]][e]), zeroed for PAD
__global__ void k_embed(const float* __restrict__ embW, const int* __restrict__ x,
                        uint16_t* __restrict__ dst) {
  int row = blockIdx.x;
  int t = row / BDIM, b = row % BDIM;
  int tok = x[b * TDIM + t];
  const float* src = embW + (size_t)tok * EDIM;
  uint16_t* d = dst + (size_t)row * EDIM;
  for (int e = threadIdx.x; e < EDIM; e += blockDim.x)
    d[e] = (tok == PADTOK) ? (uint16_t)0 : f2bf(src[e]);
}

// ---------------- GEMM kernels ----------------
// plain: C[m][n] f32 = A(bf16, MxK) · B(bf16, NxK)ᵀ   (used for gi, N=6144, K=1024)
__global__ void k_gemm_store(const uint16_t* __restrict__ A, const uint16_t* __restrict__ Bw,
                             float* __restrict__ C, int K, int Ncols) {
  int lane = threadIdx.x & 31, wave = threadIdx.x >> 5;
  int r = lane & 15, h = lane >> 4;
  int m0 = blockIdx.y * 32;
  int n0base = blockIdx.x * 512 + wave * 64;
  v8f acc[4][2];
  gemm_core<4>(A, Bw, K, Ncols, m0, n0base, acc);
#pragma unroll
  for (int j = 0; j < 4; ++j) {
    int n = n0base + j * 16 + r;
    if (n < Ncols)
#pragma unroll
      for (int sub = 0; sub < 2; ++sub)
#pragma unroll
        for (int i = 0; i < 8; ++i) {
          int m = m0 + sub * 16 + i + 8 * h;
          C[(size_t)m * Ncols + n] = acc[j][sub][i];
        }
  }
}

// lin1: ses = tanh(ses_enc · lin1_Wᵀ + b); init both hcat buffers: [:, :H]=ses (h0), [:, H:]=ses
__global__ void k_lin1(const uint16_t* __restrict__ A, const uint16_t* __restrict__ Bw,
                       const float* __restrict__ bias,
                       uint16_t* __restrict__ hcatA, uint16_t* __restrict__ hcatB) {
  int lane = threadIdx.x & 31, wave = threadIdx.x >> 5;
  int r = lane & 15, h = lane >> 4;
  int m0 = blockIdx.y * 32;
  int n0base = blockIdx.x * 512 + wave * 64;
  v8f acc[4][2];
  gemm_core<4>(A, Bw, SHDIM, HDIM, m0, n0base, acc);
#pragma unroll
  for (int j = 0; j < 4; ++j) {
    int n = n0base + j * 16 + r;
    if (n < HDIM)
#pragma unroll
      for (int sub = 0; sub < 2; ++sub)
#pragma unroll
        for (int i = 0; i < 8; ++i) {
          int m = m0 + sub * 16 + i + 8 * h;
          uint16_t hb = f2bf(tanhf(acc[j][sub][i] + bias[n]));
          hcatA[(size_t)m * H2 + n] = hb;          // h0 = ses
          hcatA[(size_t)m * H2 + HDIM + n] = hb;   // ses half (even steps read A)
          hcatB[(size_t)m * H2 + HDIM + n] = hb;   // ses half (odd steps read B)
        }
  }
}

// fused GRU step: gh GEMM for all 3 gates + gate math; writes h_full(bf16) and next hcat[:, :H]
__global__ void k_step(const uint16_t* __restrict__ hcat_rd, const uint16_t* __restrict__ Whh,
                       const float* __restrict__ gi, const float* __restrict__ bih,
                       const float* __restrict__ bhh,
                       uint16_t* __restrict__ hfull, uint16_t* __restrict__ hcat_wr, int t) {
  int lane = threadIdx.x & 31, wave = threadIdx.x >> 5;
  int r = lane & 15, h = lane >> 4;
  int m0 = blockIdx.y * 32;
  int n0 = (blockIdx.x * 8 + wave) * 16;   // gate-local column tile
  v8f acc[3][2];
#pragma unroll
  for (int g = 0; g < 3; ++g) { acc[g][0] = vzero8(); acc[g][1] = vzero8(); }
#pragma unroll 2
  for (int k0 = 0; k0 < H2; k0 += 32) {
    v16bf a0 = frag_a(hcat_rd, H2, m0, k0);
    v16bf a1 = frag_a(hcat_rd, H2, m0 + 16, k0);
#pragma unroll
    for (int g = 0; g < 3; ++g) {
      v16bf b = frag_b(Whh, H2, g * H2 + n0, k0);
      acc[g][0] = wmma_bf16(a0, b, acc[g][0]);
      acc[g][1] = wmma_bf16(a1, b, acc[g][1]);
    }
  }
#pragma unroll
  for (int sub = 0; sub < 2; ++sub)
#pragma unroll
    for (int i = 0; i < 8; ++i) {
      int m = m0 + sub * 16 + i + 8 * h;      // batch index
      int n = n0 + r;                         // gate column
      size_t row = (size_t)t * BDIM + m;
      const float* grow = gi + row * (size_t)G3;
      float gr  = acc[0][sub][i] + grow[n]           + bih[n]           + bhh[n];
      float gz  = acc[1][sub][i] + grow[H2 + n]      + bih[H2 + n]      + bhh[H2 + n];
      float hnv = acc[2][sub][i] + bhh[2 * H2 + n];
      float inv = grow[2 * H2 + n] + bih[2 * H2 + n];
      float rg = 1.0f / (1.0f + expf(-gr));
      float zg = 1.0f / (1.0f + expf(-gz));
      float nn = tanhf(inv + rg * hnv);
      float hc = bf2f(hcat_rd[(size_t)m * H2 + n]);
      float hf = (1.0f - zg) * nn + zg * hc;
      hfull[row * H2 + n] = f2bf(hf);
      if (n < HDIM) hcat_wr[(size_t)m * H2 + n] = f2bf(hf);  // h_next
    }
}

// hid_o = h_full · lin2ᵀ + b + emb  -> bf16 (batched over all t)
__global__ void k_hido(const uint16_t* __restrict__ A, const uint16_t* __restrict__ Bw,
                       const float* __restrict__ bias, const uint16_t* __restrict__ embbf,
                       uint16_t* __restrict__ hido) {
  int lane = threadIdx.x & 31, wave = threadIdx.x >> 5;
  int r = lane & 15, h = lane >> 4;
  int m0 = blockIdx.y * 32;
  int n0base = blockIdx.x * 512 + wave * 64;
  v8f acc[4][2];
  gemm_core<4>(A, Bw, H2, EDIM, m0, n0base, acc);
#pragma unroll
  for (int j = 0; j < 4; ++j) {
    int n = n0base + j * 16 + r;
    if (n < EDIM)
#pragma unroll
      for (int sub = 0; sub < 2; ++sub)
#pragma unroll
        for (int i = 0; i < 8; ++i) {
          int m = m0 + sub * 16 + i + 8 * h;
          float v = acc[j][sub][i] + bias[n] + bf2f(embbf[(size_t)m * EDIM + n]);
          hido[(size_t)m * EDIM + n] = f2bf(v);
        }
  }
}

// logits = hid_o · out_Wᵀ ; write f32 straight into d_out in (b,t,v) layout
__global__ void k_logits(const uint16_t* __restrict__ A, const uint16_t* __restrict__ Bw,
                         float* __restrict__ out) {
  int lane = threadIdx.x & 31, wave = threadIdx.x >> 5;
  int r = lane & 15, h = lane >> 4;
  int m0 = blockIdx.y * 32;
  int n0base = blockIdx.x * 512 + wave * 64;
  v8f acc[4][2];
  gemm_core<4>(A, Bw, EDIM, VSZ, m0, n0base, acc);
#pragma unroll
  for (int j = 0; j < 4; ++j) {
    int n = n0base + j * 16 + r;
    if (n < VSZ)
#pragma unroll
      for (int sub = 0; sub < 2; ++sub)
#pragma unroll
        for (int i = 0; i < 8; ++i) {
          int m = m0 + sub * 16 + i + 8 * h;     // row = t*64 + b
          int t = m >> 6, b = m & 63;
          out[((size_t)(b * TDIM + t)) * VSZ + n] = acc[j][sub][i];
        }
  }
}

// per-(t,b) log-softmax gather: ll[row] = logits[nxt] - logsumexp  (0 if last step)
__global__ void k_ll(const float* __restrict__ out, const int* __restrict__ x,
                     float* __restrict__ ll) {
  __shared__ float red[256];
  int row = blockIdx.x;                  // row = t*64 + b
  int t = row / BDIM, b = row % BDIM;
  const float* p = out + ((size_t)(b * TDIM + t)) * VSZ;
  float mx = -1e30f;
  for (int v = threadIdx.x; v < VSZ; v += 256) mx = fmaxf(mx, p[v]);
  red[threadIdx.x] = mx; __syncthreads();
  for (int s = 128; s > 0; s >>= 1) {
    if ((int)threadIdx.x < s) red[threadIdx.x] = fmaxf(red[threadIdx.x], red[threadIdx.x + s]);
    __syncthreads();
  }
  mx = red[0]; __syncthreads();
  float sum = 0.0f;
  for (int v = threadIdx.x; v < VSZ; v += 256) sum += expf(p[v] - mx);
  red[threadIdx.x] = sum; __syncthreads();
  for (int s = 128; s > 0; s >>= 1) {
    if ((int)threadIdx.x < s) red[threadIdx.x] += red[threadIdx.x + s];
    __syncthreads();
  }
  if (threadIdx.x == 0) {
    float val = 0.0f;
    if (t < TDIM - 1) {
      int nxt = x[b * TDIM + t + 1];
      val = p[nxt] - (mx + logf(red[0]));
    }
    ll[row] = val;
  }
}

__global__ void k_sum(const float* __restrict__ ll, float* __restrict__ outp) {
  int b = threadIdx.x;
  if (b < BDIM) {
    float s = 0.0f;
    for (int t = 0; t < TDIM; ++t) s += ll[t * BDIM + b];
    outp[b] = s;
  }
}

extern "C" void kernel_launch(void* const* d_in, const int* in_sizes, int n_in,
                              void* d_out, int out_size, void* d_ws, size_t ws_size,
                              hipStream_t stream) {
  (void)in_sizes; (void)n_in; (void)out_size; (void)ws_size;
  const float* ses   = (const float*)d_in[0];
  const int*   x     = (const int*)  d_in[1];
  const float* embW  = (const float*)d_in[2];
  const float* Wih   = (const float*)d_in[3];
  const float* Whh   = (const float*)d_in[4];
  const float* bih   = (const float*)d_in[5];
  const float* bhh   = (const float*)d_in[6];
  const float* lin1W = (const float*)d_in[7];
  const float* lin1b = (const float*)d_in[8];
  const float* lin2W = (const float*)d_in[9];
  const float* lin2b = (const float*)d_in[10];
  const float* outW  = (const float*)d_in[11];
  float* out = (float*)d_out;

  char* ws = (char*)d_ws;
  size_t off = 0;
  auto alloc = [&](size_t bytes) -> void* {
    void* p = ws + off;
    off = (off + bytes + 255) & ~(size_t)255;
    return p;
  };
  uint16_t* ses_bf  = (uint16_t*)alloc((size_t)BDIM * SHDIM * 2);
  uint16_t* l1W_bf  = (uint16_t*)alloc((size_t)HDIM * SHDIM * 2);
  uint16_t* Wih_bf  = (uint16_t*)alloc((size_t)G3 * EDIM * 2);
  uint16_t* Whh_bf  = (uint16_t*)alloc((size_t)G3 * H2 * 2);
  uint16_t* l2W_bf  = (uint16_t*)alloc((size_t)EDIM * H2 * 2);
  uint16_t* oW_bf   = (uint16_t*)alloc((size_t)VSZ * EDIM * 2);
  uint16_t* emb_bf  = (uint16_t*)alloc((size_t)ROWS * EDIM * 2);
  uint16_t* hcatA   = (uint16_t*)alloc((size_t)BDIM * H2 * 2);
  uint16_t* hcatB   = (uint16_t*)alloc((size_t)BDIM * H2 * 2);
  float*    gi      = (float*)   alloc((size_t)ROWS * G3 * 4);
  uint16_t* hfull   = (uint16_t*)alloc((size_t)ROWS * H2 * 2);
  uint16_t* hido    = (uint16_t*)alloc((size_t)ROWS * EDIM * 2);
  float*    ll      = (float*)   alloc((size_t)ROWS * 4);

  // 1) fp32 -> bf16 conversions (weights + ses input)
  k_convert<<<2048, 256, 0, stream>>>(ses,   ses_bf, (size_t)BDIM * SHDIM);
  k_convert<<<2048, 256, 0, stream>>>(lin1W, l1W_bf, (size_t)HDIM * SHDIM);
  k_convert<<<4096, 256, 0, stream>>>(Wih,   Wih_bf, (size_t)G3 * EDIM);
  k_convert<<<4096, 256, 0, stream>>>(Whh,   Whh_bf, (size_t)G3 * H2);
  k_convert<<<2048, 256, 0, stream>>>(lin2W, l2W_bf, (size_t)EDIM * H2);
  k_convert<<<8192, 256, 0, stream>>>(outW,  oW_bf,  (size_t)VSZ * EDIM);

  // 2) embedding gather (all t, PAD-zeroed)
  k_embed<<<ROWS, 256, 0, stream>>>(embW, x, emb_bf);

  // 3) ses = tanh(lin1), seeds both hcat ping-pong buffers (h0 = ses)
  k_lin1<<<dim3(2, 2), 256, 0, stream>>>(ses_bf, l1W_bf, lin1b, hcatA, hcatB);

  // 4) gi for ALL timesteps in one GEMM: (3200 x 6144 x 1024)
  k_gemm_store<<<dim3(12, ROWS / 32), 256, 0, stream>>>(emb_bf, Wih_bf, gi, EDIM, G3);

  // 5) sequential recurrence: only the Whh GEMM + gates are on the chain
  for (int t = 0; t < TDIM; ++t) {
    const uint16_t* rd = (t & 1) ? hcatB : hcatA;
    uint16_t*       wr = (t & 1) ? hcatA : hcatB;
    k_step<<<dim3(16, 2), 256, 0, stream>>>(rd, Whh_bf, gi, bih, bhh, hfull, wr, t);
  }

  // 6) hid_o for all t: (3200 x 1024 x 2048)
  k_hido<<<dim3(2, ROWS / 32), 256, 0, stream>>>(hfull, l2W_bf, lin2b, emb_bf, hido);

  // 7) logits for all t: (3200 x 30000 x 1024) -> d_out (b,t,v)
  k_logits<<<dim3(59, ROWS / 32), 256, 0, stream>>>(hido, oW_bf, out);

  // 8) log-softmax gather + per-batch sum -> d_out tail
  k_ll<<<ROWS, 256, 0, stream>>>(out, x, ll);
  k_sum<<<1, 64, 0, stream>>>(ll, out + (size_t)BDIM * TDIM * VSZ);
}
